// VELMBackbone_6528350290248
// MI455X (gfx1250) — matrix operand order repaired and checked by reading, not compiled, not run on hardware
//
#include <hip/hip_runtime.h>
#include <hip/hip_bf16.h>
#include <cstdint>
#include <cstddef>

// ---------------------------------------------------------------------------
// VELM backbone for MI455X (gfx1250, wave32, WMMA).
//  - GEMMs: bf16 v_wmma_f32_16x16x32_bf16, f32 accum, fused SiLU/sigmoid/
//    residual epilogues, WGP-scope prefetch of next k-tile.
//  - Scan: row-parallel recurrence; state lives in VGPRs; k_{t+1} is
//    double-buffered into LDS with global_load_async_to_lds_b32 (ASYNCcnt)
//    so the sequential critical path never waits on global memory.
//  - q-branch of the reference is dead code (unused in readout) -> skipped.
// ---------------------------------------------------------------------------

#define TT    1024
#define DD    512
#define HDD   32
#define LRR   32
#define FFND  2048
#define CONVK 4

typedef __attribute__((ext_vector_type(16))) __bf16          v16bf;
typedef __attribute__((ext_vector_type(8)))  float           v8f;
typedef __attribute__((ext_vector_type(8)))  unsigned short  us8;

__device__ __forceinline__ unsigned short f2bf(float f) {
  union { float f; unsigned int u; } c; c.f = f;
  unsigned int u = c.u + (0x7FFFu + ((c.u >> 16) & 1u));   // round-nearest-even
  return (unsigned short)(u >> 16);
}

// ---------------------------- elementwise ----------------------------------

__global__ void cvt_bf16_kernel(const float* __restrict__ s,
                                unsigned short* __restrict__ d, int n) {
  int i = blockIdx.x * blockDim.x + threadIdx.x;
  if (i < n) d[i] = f2bf(s[i]);
}

__global__ void mul_to_bf16_kernel(const float* __restrict__ a,
                                   const float* __restrict__ b,
                                   unsigned short* __restrict__ d, int n) {
  int i = blockIdx.x * blockDim.x + threadIdx.x;
  if (i < n) d[i] = f2bf(a[i] * b[i]);
}

// ------------------------------ RMSNorm ------------------------------------
// one row (D=512) per 256-thread block; 2 elements/thread; wave32 + LDS reduce
__global__ void __launch_bounds__(256)
rmsnorm_kernel(const float* __restrict__ x, const float* __restrict__ w,
               float* __restrict__ of32, unsigned short* __restrict__ obf,
               int has_f32) {
  __shared__ float red[8];
  __shared__ float scale_s;
  int row = blockIdx.x;
  int tid = threadIdx.x;
  const float* xr = x + (size_t)row * DD;
  float a = xr[tid], b = xr[tid + 256];
  float s = a * a + b * b;
#pragma unroll
  for (int m = 16; m >= 1; m >>= 1) s += __shfl_xor(s, m, 32);
  if ((tid & 31) == 0) red[tid >> 5] = s;
  __syncthreads();
  if (tid == 0) {
    float tot = 0.f;
#pragma unroll
    for (int i = 0; i < 8; ++i) tot += red[i];
    scale_s = rsqrtf(tot / (float)DD + 1e-6f);
  }
  __syncthreads();
  float sc = scale_s;
  float o0 = a * sc * w[tid];
  float o1 = b * sc * w[tid + 256];
  size_t base = (size_t)row * DD;
  if (has_f32) { of32[base + tid] = o0; of32[base + tid + 256] = o1; }
  obf[base + tid] = f2bf(o0);
  obf[base + tid + 256] = f2bf(o1);
}

// ------------------- causal depthwise conv + head RMSNorm ------------------
// block = one timestep (512 threads = 16 waves); each wave == one head (HD=32)
__global__ void __launch_bounds__(512)
conv_headnorm_kernel(const float* __restrict__ kpre, const float* __restrict__ vpre,
                     const float* __restrict__ ckw, const float* __restrict__ ckb,
                     const float* __restrict__ cvw, const float* __restrict__ cvb,
                     const float* __restrict__ knw,
                     float* __restrict__ kseq, float* __restrict__ vseq) {
  int t = blockIdx.x, c = threadIdx.x;
  float kv = ckb[c], vv = cvb[c];
#pragma unroll
  for (int j = 0; j < CONVK; ++j) {
    int tt = t + j - (CONVK - 1);
    if (tt >= 0) {
      kv += kpre[(size_t)tt * DD + c] * ckw[c * CONVK + j];
      vv += vpre[(size_t)tt * DD + c] * cvw[c * CONVK + j];
    }
  }
  float ss = kv * kv;                      // per-head sum over the 32-lane wave
#pragma unroll
  for (int m = 16; m >= 1; m >>= 1) ss += __shfl_xor(ss, m, 32);
  float inv = rsqrtf(ss / (float)HDD + 1e-6f);
  kseq[(size_t)t * DD + c] = kv * inv * knw[c & (HDD - 1)];
  vseq[(size_t)t * DD + c] = vv;
}

// ------------------------------- scan --------------------------------------
// 32 blocks x 512 threads. Block owns 16 state rows; a 32-lane wave owns one
// row; each lane holds 16 state columns in VGPRs for the whole sequence.
// k_{t+1} is prefetched into a double-buffered LDS stage with
// global_load_async_to_lds_b32 while step t computes (ASYNCcnt pipelining).
__global__ void __launch_bounds__(512)
scan_kernel(const float* __restrict__ kseq, const float* __restrict__ vseq,
            const float* __restrict__ eta, const float* __restrict__ alpha,
            float* __restrict__ out) {
  __shared__ float sk[2][DD];       // only LDS object in this kernel -> base 0
  int tid = threadIdx.x;
  int p = tid & 31;                 // lane within row-group
  int rowLocal = tid >> 5;          // 0..15
  int row = blockIdx.x * 16 + rowLocal;
  int c0 = p * 16;
  unsigned lds0 = (unsigned)(tid * 4);            // LDS byte offsets of sk[0/1]
  unsigned lds1 = (unsigned)((DD + tid) * 4);

  // prologue: stage k_0 into buffer 0 (one b32 per lane, async DMA to LDS)
  {
    const float* g0 = kseq + tid;
    asm volatile("global_load_async_to_lds_b32 %0, %1, off"
                 :: "v"(lds0), "v"(g0) : "memory");
  }

  float S[16];
#pragma unroll
  for (int j = 0; j < 16; ++j) S[j] = 0.f;

  for (int t = 0; t < TT; ++t) {
    int b = t & 1;
    if (t + 1 < TT) {
      // issue prefetch of k_{t+1} into the other buffer; its last readers
      // finished before the trailing barrier of iteration t-1.
      const float* gn = kseq + (size_t)(t + 1) * DD + tid;
      unsigned ldsn = b ? lds0 : lds1;
      asm volatile("global_load_async_to_lds_b32 %0, %1, off"
                   :: "v"(ldsn), "v"(gn) : "memory");
      asm volatile("s_wait_asynccnt 0x1" ::: "memory");  // retire k_t copy
    } else {
      asm volatile("s_wait_asynccnt 0x0" ::: "memory");
    }
    __syncthreads();                 // every wave's slice of k_t is in LDS

    float kk[16];
    const float4* kp = (const float4*)&sk[b][c0];        // 64B-aligned b128
#pragma unroll
    for (int q = 0; q < 4; ++q) *(float4*)&kk[q * 4] = kp[q];

    float vr = vseq[(size_t)t * DD + row];
    float er = eta[(size_t)t * DD + row];
    float ar = alpha[(size_t)t * DD + row];

    float d = 0.f;                                       // error = S@k - v
#pragma unroll
    for (int j = 0; j < 16; ++j) d = fmaf(S[j], kk[j], d);
#pragma unroll
    for (int m = 16; m >= 1; m >>= 1) d += __shfl_xor(d, m, 32);
    float ce = er * (d - vr);

    float d2 = 0.f;                                      // update + out = S'@k
#pragma unroll
    for (int j = 0; j < 16; ++j) {
      float ns = ar * S[j] - ce * kk[j];
      ns = fminf(10.f, fmaxf(-10.f, ns));
      S[j] = ns;
      d2 = fmaf(ns, kk[j], d2);
    }
#pragma unroll
    for (int m = 16; m >= 1; m >>= 1) d2 += __shfl_xor(d2, m, 32);
    if (p == 0) out[(size_t)t * DD + row] = d2;
    __syncthreads();                 // buffer b fully consumed
  }
}

// ---------------------------- WMMA GEMM ------------------------------------
// C[M,N] (f32, row-major) = A[M,K] (bf16, row-major) x W[N,K]^T (bf16, rowmaj)
// block = 256 threads = 8 waves; tile 64(M) x 64(N); wave -> 16x32 output.
// EPI: 0 = none, 1 = SiLU, 2 = sigmoid, 3 = add residual R.
template <int EPI>
__global__ void __launch_bounds__(256)
wmma_gemm_bf16_kernel(const unsigned short* __restrict__ A,
                      const unsigned short* __restrict__ W,
                      const float* __restrict__ R,
                      float* __restrict__ C, int M, int N, int K) {
  int tid = threadIdx.x;
  int wid = tid >> 5, lane = tid & 31;
  int l15 = lane & 15, hi = lane >> 4;
  int m0 = blockIdx.x * 64 + (wid & 3) * 16;
  int n0 = blockIdx.y * 64 + (wid >> 2) * 32;

  // A fragment (16x32 bf16): lane = M row; lane halves hold K 0-15 / 16-31.
  const unsigned short* arow = A + (size_t)(m0 + l15) * K;
  int ak = hi ? 16 : 0;

  // B fragments (32x16): lane&15 = column n; lane halves hold K 0-15 / 16-31.
  int nA = n0 + l15, nB = n0 + 16 + l15;
  bool okA = nA < N, okB = nB < N;
  const unsigned short* wrowA = W + (size_t)(okA ? nA : 0) * K;
  const unsigned short* wrowB = W + (size_t)(okB ? nB : 0) * K;
  int wk = hi ? 16 : 0;

  v8f acc0 = {}, acc1 = {};
  for (int k0 = 0; k0 < K; k0 += 32) {
    union { v16bf v; us8 h[2]; } ua, ub0, ub1;
    ua.h[0]  = *(const us8*)(arow  + k0 + ak);
    ua.h[1]  = *(const us8*)(arow  + k0 + 8 + ak);
    ub0.h[0] = *(const us8*)(wrowA + k0 + wk);
    ub0.h[1] = *(const us8*)(wrowA + k0 + wk + 8);
    ub1.h[0] = *(const us8*)(wrowB + k0 + wk);
    ub1.h[1] = *(const us8*)(wrowB + k0 + wk + 8);
    if (k0 + 32 < K) {               // WGP-scope prefetch of next k-tile
      __builtin_prefetch(arow  + k0 + 32, 0, 3);
      __builtin_prefetch(wrowA + k0 + 32, 0, 3);
      __builtin_prefetch(wrowB + k0 + 32, 0, 3);
    }
    acc0 = __builtin_amdgcn_wmma_f32_16x16x32_bf16(false, ua.v, false, ub0.v,
                                                   (short)0, acc0, false, false);
    acc1 = __builtin_amdgcn_wmma_f32_16x16x32_bf16(false, ua.v, false, ub1.v,
                                                   (short)0, acc1, false, false);
  }

  int mbase = m0 + (hi ? 8 : 0);
#pragma unroll
  for (int r = 0; r < 8; ++r) {
    if (okA) {
      size_t idx = (size_t)(mbase + r) * N + nA;
      float v0 = acc0[r];
      if (EPI == 1) v0 = v0 / (1.f + __expf(-v0));
      if (EPI == 2) v0 = 1.f / (1.f + __expf(-v0));
      if (EPI == 3) v0 += R[idx];
      C[idx] = v0;
    }
    if (okB) {
      size_t idx = (size_t)(mbase + r) * N + nB;
      float v1 = acc1[r];
      if (EPI == 1) v1 = v1 / (1.f + __expf(-v1));
      if (EPI == 2) v1 = 1.f / (1.f + __expf(-v1));
      if (EPI == 3) v1 += R[idx];
      C[idx] = v1;
    }
  }
}

// ------------------------------ host side ----------------------------------

static inline void launch_gemm(int epi, const unsigned short* A,
                               const unsigned short* W, const float* R,
                               float* C, int M, int N, int K, hipStream_t s) {
  dim3 g(M / 64, (N + 63) / 64), b(256);
  switch (epi) {
    case 1:  wmma_gemm_bf16_kernel<1><<<g, b, 0, s>>>(A, W, R, C, M, N, K); break;
    case 2:  wmma_gemm_bf16_kernel<2><<<g, b, 0, s>>>(A, W, R, C, M, N, K); break;
    case 3:  wmma_gemm_bf16_kernel<3><<<g, b, 0, s>>>(A, W, R, C, M, N, K); break;
    default: wmma_gemm_bf16_kernel<0><<<g, b, 0, s>>>(A, W, R, C, M, N, K); break;
  }
}

extern "C" void kernel_launch(void* const* d_in, const int* in_sizes, int n_in,
                              void* d_out, int out_size, void* d_ws, size_t ws_size,
                              hipStream_t stream) {
  (void)in_sizes; (void)n_in; (void)out_size; (void)ws_size;

  const float* x        = (const float*)d_in[0];
  const float* wk_w     = (const float*)d_in[2];
  const float* wv_w     = (const float*)d_in[3];
  const float* wo_w     = (const float*)d_in[4];
  const float* convk_w  = (const float*)d_in[7];
  const float* convk_b  = (const float*)d_in[8];
  const float* convv_w  = (const float*)d_in[9];
  const float* convv_b  = (const float*)d_in[10];
  const float* kn_w     = (const float*)d_in[12];
  const float* eta_down = (const float*)d_in[13];
  const float* eta_up   = (const float*)d_in[14];
  const float* al_down  = (const float*)d_in[15];
  const float* al_up    = (const float*)d_in[16];
  const float* gate_w   = (const float*)d_in[17];
  const float* norm1_w  = (const float*)d_in[18];
  const float* norm2_w  = (const float*)d_in[19];
  const float* ffn_g    = (const float*)d_in[20];
  const float* ffn_u    = (const float*)d_in[21];
  const float* ffn_d    = (const float*)d_in[22];

  char* ws = (char*)d_ws;
  size_t off = 0;
  auto alloc = [&](size_t bytes) -> void* {
    void* p = ws + off;
    off = (off + bytes + 255) & ~(size_t)255;
    return p;
  };
  const size_t TD = (size_t)TT * DD;

  unsigned short* wk_bf   = (unsigned short*)alloc(DD * DD * 2);
  unsigned short* wv_bf   = (unsigned short*)alloc(DD * DD * 2);
  unsigned short* wo_bf   = (unsigned short*)alloc(DD * DD * 2);
  unsigned short* gw_bf   = (unsigned short*)alloc(DD * DD * 2);
  unsigned short* etad_bf = (unsigned short*)alloc(LRR * DD * 2);
  unsigned short* etau_bf = (unsigned short*)alloc(DD * LRR * 2);
  unsigned short* ald_bf  = (unsigned short*)alloc(LRR * DD * 2);
  unsigned short* alu_bf  = (unsigned short*)alloc(DD * LRR * 2);
  unsigned short* ffng_bf = (unsigned short*)alloc((size_t)FFND * DD * 2);
  unsigned short* ffnu_bf = (unsigned short*)alloc((size_t)FFND * DD * 2);
  unsigned short* ffnd_bf = (unsigned short*)alloc((size_t)DD * FFND * 2);

  float*          n1_f32  = (float*)alloc(TD * 4);
  unsigned short* n1_bf   = (unsigned short*)alloc(TD * 2);
  float*          kpre    = (float*)alloc(TD * 4);
  float*          vpre    = (float*)alloc(TD * 4);
  float*          gate    = (float*)alloc(TD * 4);
  float*          eta_mid = (float*)alloc((size_t)TT * LRR * 4);
  unsigned short* etam_bf = (unsigned short*)alloc((size_t)TT * LRR * 2);
  float*          eta     = (float*)alloc(TD * 4);
  float*          al_mid  = (float*)alloc((size_t)TT * LRR * 4);
  unsigned short* alm_bf  = (unsigned short*)alloc((size_t)TT * LRR * 2);
  float*          alpha   = (float*)alloc(TD * 4);
  float*          kseq    = (float*)alloc(TD * 4);
  float*          vseq    = (float*)alloc(TD * 4);
  float*          scano   = (float*)alloc(TD * 4);
  unsigned short* prod_bf = (unsigned short*)alloc(TD * 2);
  float*          x1      = (float*)alloc(TD * 4);
  unsigned short* n2_bf   = (unsigned short*)alloc(TD * 2);
  float*          G       = (float*)alloc((size_t)TT * FFND * 4);
  float*          U       = (float*)alloc((size_t)TT * FFND * 4);
  unsigned short* h_bf    = (unsigned short*)alloc((size_t)TT * FFND * 2);

  auto cvt = [&](const float* s, unsigned short* d, int n) {
    cvt_bf16_kernel<<<(n + 255) / 256, 256, 0, stream>>>(s, d, n);
  };

  // 1) weight conversion f32 -> bf16 (tiny; everything stays L2-resident)
  cvt(wk_w,     wk_bf,   DD * DD);
  cvt(wv_w,     wv_bf,   DD * DD);
  cvt(wo_w,     wo_bf,   DD * DD);
  cvt(gate_w,   gw_bf,   DD * DD);
  cvt(eta_down, etad_bf, LRR * DD);
  cvt(eta_up,   etau_bf, DD * LRR);
  cvt(al_down,  ald_bf,  LRR * DD);
  cvt(al_up,    alu_bf,  DD * LRR);
  cvt(ffn_g,    ffng_bf, FFND * DD);
  cvt(ffn_u,    ffnu_bf, FFND * DD);
  cvt(ffn_d,    ffnd_bf, DD * FFND);

  // 2) norm1
  rmsnorm_kernel<<<TT, 256, 0, stream>>>(x, norm1_w, n1_f32, n1_bf, 1);

  // 3) projections (q-branch skipped: dead in the reference readout)
  launch_gemm(0, n1_bf, wk_bf, nullptr, kpre, TT, DD, DD, stream);
  launch_gemm(0, n1_bf, wv_bf, nullptr, vpre, TT, DD, DD, stream);
  launch_gemm(1, n1_bf, gw_bf, nullptr, gate, TT, DD, DD, stream);  // SiLU

  // 4) low-rank eta / alpha gates (sigmoid fused in second stage)
  launch_gemm(0, n1_bf, etad_bf, nullptr, eta_mid, TT, LRR, DD, stream);
  cvt(eta_mid, etam_bf, TT * LRR);
  launch_gemm(2, etam_bf, etau_bf, nullptr, eta, TT, DD, LRR, stream);
  launch_gemm(0, n1_bf, ald_bf, nullptr, al_mid, TT, LRR, DD, stream);
  cvt(al_mid, alm_bf, TT * LRR);
  launch_gemm(2, alm_bf, alu_bf, nullptr, alpha, TT, DD, LRR, stream);

  // 5) causal depthwise conv + per-head RMSNorm (k only; v passthrough)
  conv_headnorm_kernel<<<TT, 512, 0, stream>>>(kpre, vpre, convk_w, convk_b,
                                               convv_w, convv_b, kn_w,
                                               kseq, vseq);

  // 6) sequential scan: state in VGPRs, async double-buffered k_t in LDS
  scan_kernel<<<DD / 16, 512, 0, stream>>>(kseq, vseq, eta, alpha, scano);

  // 7) gated output projection, residual fused into GEMM epilogue
  mul_to_bf16_kernel<<<(int)((TD + 255) / 256), 256, 0, stream>>>(scano, gate,
                                                                  prod_bf, (int)TD);
  launch_gemm(3, prod_bf, wo_bf, x, x1, TT, DD, DD, stream);   // x1 = x + out@wo^T

  // 8) FFN (SwiGLU), final residual fused into last GEMM epilogue
  rmsnorm_kernel<<<TT, 256, 0, stream>>>(x1, norm2_w, nullptr, n2_bf, 0);
  launch_gemm(1, n2_bf, ffng_bf, nullptr, G, TT, FFND, DD, stream);  // SiLU
  launch_gemm(0, n2_bf, ffnu_bf, nullptr, U, TT, FFND, DD, stream);
  int nf = TT * FFND;
  mul_to_bf16_kernel<<<(nf + 255) / 256, 256, 0, stream>>>(G, U, h_bf, nf);
  launch_gemm(3, h_bf, ffnd_bf, x1, (float*)d_out, TT, DD, FFND, stream);
}
